// TP_lstm_46866683134646
// MI455X (gfx1250) — matrix operand order repaired and checked by reading, not compiled, not run on hardware
//
#include <hip/hip_runtime.h>
#include <hip/hip_bf16.h>

// ---------------------------------------------------------------------------
// TP-LSTM forward for MI455X (gfx1250, wave32, WMMA).
// One workgroup per batch (B=128 WGs, 256 thr = 8 waves). Wave w owns rows
// s in [16w,16w+16). h state lives in LDS (f16); all three per-step GEMMs run
// on v_wmma_f32_16x16x32_f16 with f32 accumulation; c state stays in VGPRs.
// ---------------------------------------------------------------------------

#define Bb     128
#define Ss     128
#define Tt     256
#define Ii     3
#define Hh     64
#define TPRED  4

typedef __attribute__((ext_vector_type(16))) _Float16 v16h;
typedef __attribute__((ext_vector_type(8)))  float    v8f;
typedef __attribute__((ext_vector_type(4)))  float    f4;

#define SA 104   // A-region row stride in halfs: 208B = 13*16B -> aligned + bank-spread

// LDS layout (byte offsets)
#define OFF_BG   0        // gates B' frags: 48 frags * 512 halfs  (K=96, N=256)
#define OFF_BS   49152    // spatial-gate B:  16 frags * 512 halfs  (K=128, N=64)
#define OFF_BO   65536    // out-head B:       4 frags * 512 halfs  (K=64,  N=32)
#define OFF_A    69632    // A region: 128 rows * 104 halfs (cols 0..63 = h, 64..66 = x*lc, rest 0)
#define OFF_ZR   96256    // zero row (64 halfs, padded to 128B)
#define OFF_GB   96384    // gate bias f32[256] = b_ih + b_hh
#define OFF_SB   97408    // se_b  f32[64]
#define OFF_OB   97664    // out_b1 f32[32]
#define OFF_W2   97792    // out_w2 f32[32]
#define OFF_SCR  97920    // out-head scratch: 8 waves * 16 rows * 36 f32
#define SMEM_BYTES (97920 + 8*16*36*4)   // 116352 B -> 2 WGs / 320KB WGP

static __device__ __forceinline__ float sigf(float x)     { return 1.f / (1.f + __expf(-x)); }
static __device__ __forceinline__ float tanhfast(float x) { return 2.f / (1.f + __expf(-2.f * x)) - 1.f; }
static __device__ __forceinline__ v8f   splat8(float x)   { v8f r = {x,x,x,x,x,x,x,x}; return r; }

// B fragment: pre-swizzled, each lane's 16 halfs contiguous -> 2x ds_load_b128
static __device__ __forceinline__ v16h ldB(const _Float16* frag, int lane) {
  union { v16h h; f4 f[2]; } u;
  const f4* p = (const f4*)(frag + lane * 16);
  u.f[0] = p[0];
  u.f[1] = p[1];
  return u.h;
}

// A fragment from row-major f16 row: lanes<16 take k {0..7,16..23}, lanes>=16 {8..15,24..31}
static __device__ __forceinline__ v16h ldA(const _Float16* row, int kb, int cb) {
  union { v16h h; f4 f[2]; } u;
  u.f[0] = *(const f4*)(row + kb + cb);
  u.f[1] = *(const f4*)(row + kb + cb + 16);
  return u.h;
}

static __device__ __forceinline__ v8f wmma_f16(v16h a, v16h b, v8f c) {
  return __builtin_amdgcn_wmma_f32_16x16x32_f16(false, a, false, b, (short)0, c, false, false);
}

extern "C" __global__ __launch_bounds__(256)
void tp_lstm_kernel(const float* __restrict__ inp,   const float* __restrict__ lane_g,
                    const float* __restrict__ lg_w1, const float* __restrict__ lg_b1,
                    const float* __restrict__ lg_w2, const float* __restrict__ lg_b2,
                    const float* __restrict__ se_w,  const float* __restrict__ se_b,
                    const float* __restrict__ w_ih,  const float* __restrict__ w_hh,
                    const float* __restrict__ b_ih,  const float* __restrict__ b_hh,
                    const float* __restrict__ out_w1,const float* __restrict__ out_b1,
                    const float* __restrict__ out_w2,const float* __restrict__ out_b2,
                    float* __restrict__ out)
{
  extern __shared__ char smem[];
  _Float16* BG  = (_Float16*)(smem + OFF_BG);
  _Float16* BS  = (_Float16*)(smem + OFF_BS);
  _Float16* BO  = (_Float16*)(smem + OFF_BO);
  _Float16* Ah  = (_Float16*)(smem + OFF_A);
  _Float16* ZRp = (_Float16*)(smem + OFF_ZR);
  float*    GB  = (float*)(smem + OFF_GB);
  float*    SBl = (float*)(smem + OFF_SB);
  float*    OB  = (float*)(smem + OFF_OB);
  float*    W2  = (float*)(smem + OFF_W2);
  float*    SCR = (float*)(smem + OFF_SCR);

  const int tid  = threadIdx.x;
  const int b    = blockIdx.x;
  const int lane = tid & 31;
  const int w    = tid >> 5;
  const int s0   = w * 16;
  const int nl   = lane & 15;
  const int cb   = (lane < 16) ? 0 : 8;
  const int srow = s0 + nl;             // A row this lane supplies to WMMA
  const int mrow = (lane >> 4) * 8;     // C-layout row offset for this half-wave

  // -------------------- setup --------------------
  // Zero A region (h=0 init + x padding) and the boundary zero-row.
  {
    unsigned int* za = (unsigned int*)Ah;
    for (int i = tid; i < (Ss * SA) / 2; i += 256) za[i] = 0u;
    unsigned int* zz = (unsigned int*)ZRp;
    if (tid < 32) zz[tid] = 0u;
  }

  // Pre-swizzle all B matrices into WMMA fragment order (f16).
  // gates B'[k][n]: k<64 -> w_hh[n][k]; k in 64..66 -> w_ih[n][k-64]; else 0.
  for (int i = tid; i < 68 * 512; i += 256) {
    int f = i >> 9, r = i & 511, ln = r >> 4, j = r & 15;
    int n_l = ln & 15;
    int k_l = (j < 8) ? ((ln >> 4) * 8 + j) : (16 + (ln >> 4) * 8 + (j - 8));
    float v;
    _Float16* dst;
    if (f < 48) {
      int nt = f / 3, kt = f - nt * 3;
      int n = nt * 16 + n_l, k = kt * 32 + k_l;
      v = (k < 64) ? w_hh[n * 64 + k] : ((k < 67) ? w_ih[n * 3 + (k - 64)] : 0.f);
      dst = BG + i;
    } else if (f < 64) {
      int f2 = f - 48;
      int nt = f2 >> 2, kt = f2 & 3;
      int n = nt * 16 + n_l, k = kt * 32 + k_l;
      v = se_w[n * 128 + k];
      dst = BS + f2 * 512 + r;
    } else {
      int f3 = f - 64;
      int nt = f3 >> 1, kt = f3 & 1;
      int n = nt * 16 + n_l, k = kt * 32 + k_l;
      v = out_w1[n * 64 + k];
      dst = BO + f3 * 512 + r;
    }
    *dst = (_Float16)v;
  }

  // Biases into LDS.
  if (tid < 256) GB[tid] = b_ih[tid] + b_hh[tid];
  if (tid < 64)  SBl[tid] = se_b[tid];
  if (tid < 32)  { OB[tid] = out_b1[tid]; W2[tid] = out_w2[tid]; }

  // Lane controller: sigmoid(FC 1->32->1), per batch (computed redundantly).
  float lc;
  {
    float lv = lane_g[b];
    float o  = lg_b2[0];
    for (int j2 = 0; j2 < 32; ++j2)
      o += fmaxf(lv * lg_w1[j2] + lg_b1[j2], 0.f) * lg_w2[j2];
    lc = sigf(o);
  }
  const float ob2 = out_b2[0];

  // x(t=0) * lc into A region x-slot (cols 64..66).
  if (tid < 128) {
    const float* xp = inp + ((size_t)(b * Ss + tid) * Tt + 0) * Ii;
    _Float16* xd = Ah + tid * SA + 64;
    xd[0] = (_Float16)(xp[0] * lc);
    xd[1] = (_Float16)(xp[1] * lc);
    xd[2] = (_Float16)(xp[2] * lc);
  }
  __syncthreads();

  // Per-lane bias registers (C layout: N = lane%16 within each n-tile).
  float biasg[16];
  #pragma unroll
  for (int nt = 0; nt < 16; ++nt) biasg[nt] = GB[nt * 16 + nl];
  float biass[4];
  #pragma unroll
  for (int nt = 0; nt < 4; ++nt) biass[nt] = SBl[nt * 16 + nl];
  float biaso[2];
  #pragma unroll
  for (int nt = 0; nt < 2; ++nt) biaso[nt] = OB[nt * 16 + nl];

  // Cell state c in registers, C-fragment layout (exact fp32 across 256 steps).
  v8f cf[4];
  #pragma unroll
  for (int q = 0; q < 4; ++q) cf[q] = splat8(0.f);

  float* scrw = SCR + w * 16 * 36;
  const _Float16* rowA = Ah + srow * SA;

  // ==================== time scan ====================
  for (int t = 0; t < Tt; ++t) {
    // ---- phase A: GEMMs reading h_prev (+x_t) from A region ----
    v16h agt[3];
    #pragma unroll
    for (int kt = 0; kt < 3; ++kt) agt[kt] = ldA(rowA, kt * 32, cb);

    const _Float16* rowUp = (srow + 1 < Ss) ? (Ah + (srow + 1) * SA) : ZRp;
    const _Float16* rowDn = (srow >= 1)     ? (Ah + (srow - 1) * SA) : ZRp;
    v16h asg[4];
    asg[0] = ldA(rowUp, 0,  cb);   // sg_in k 0..31   = h_after
    asg[1] = ldA(rowUp, 32, cb);   // sg_in k 32..63  = h_after
    asg[2] = ldA(rowDn, 0,  cb);   // sg_in k 64..95  = h_before
    asg[3] = ldA(rowDn, 32, cb);   // sg_in k 96..127 = h_before

    v8f hnew[4];
    #pragma unroll
    for (int t4 = 0; t4 < 4; ++t4) {
      v8f ai = splat8(biasg[0 * 4 + t4]);
      v8f af = splat8(biasg[1 * 4 + t4]);
      v8f ag = splat8(biasg[2 * 4 + t4]);
      v8f ao = splat8(biasg[3 * 4 + t4]);
      #pragma unroll
      for (int kt = 0; kt < 3; ++kt) {
        ai = wmma_f16(agt[kt], ldB(BG + ((0 * 4 + t4) * 3 + kt) * 512, lane), ai);
        af = wmma_f16(agt[kt], ldB(BG + ((1 * 4 + t4) * 3 + kt) * 512, lane), af);
        ag = wmma_f16(agt[kt], ldB(BG + ((2 * 4 + t4) * 3 + kt) * 512, lane), ag);
        ao = wmma_f16(agt[kt], ldB(BG + ((3 * 4 + t4) * 3 + kt) * 512, lane), ao);
      }
      v8f as_ = splat8(biass[t4]);
      #pragma unroll
      for (int kt = 0; kt < 4; ++kt)
        as_ = wmma_f16(asg[kt], ldB(BS + (t4 * 4 + kt) * 512, lane), as_);

      // LSTM cell elementwise (trans ops co-execute with WMMA).
      v8f cn, hv;
      #pragma unroll
      for (int e = 0; e < 8; ++e) {
        float c2  = sigf(af[e]) * cf[t4][e] + sigf(ai[e]) * tanhfast(ag[e]);
        float hn  = sigf(ao[e]) * tanhfast(c2) * sigf(as_[e]);
        cn[e] = c2;
        hv[e] = hn;
      }
      cf[t4]   = cn;
      hnew[t4] = hv;
    }
    __syncthreads();   // barrier1: everyone done reading h_prev / x_t

    // ---- phase B: write h_new (f16) back; prefetch + stage x_{t+1} ----
    float x0 = 0.f, x1 = 0.f, x2 = 0.f;
    const bool doX = (t + 1 < Tt) && (tid < 128);   // wave-uniform branch
    if (doX) {
      const float* xp = inp + ((size_t)(b * Ss + tid) * Tt + (t + 1)) * Ii;
      x0 = xp[0]; x1 = xp[1]; x2 = xp[2];           // latency hides under ds stores
    }
    #pragma unroll
    for (int t4 = 0; t4 < 4; ++t4)
      #pragma unroll
      for (int e = 0; e < 8; ++e)
        Ah[(s0 + mrow + e) * SA + t4 * 16 + nl] = (_Float16)hnew[t4][e];
    if (doX) {
      _Float16* xd = Ah + tid * SA + 64;
      xd[0] = (_Float16)(x0 * lc);
      xd[1] = (_Float16)(x1 * lc);
      xd[2] = (_Float16)(x2 * lc);
    }
    __syncthreads();   // barrier2: h_new visible WG-wide

    // ---- phase C: output head  relu(h_new @ W1^T + b1) @ W2^T + b2 ----
    v16h a0 = ldA(rowA, 0,  cb);
    v16h a1 = ldA(rowA, 32, cb);
    #pragma unroll
    for (int nt = 0; nt < 2; ++nt) {
      v8f oa = splat8(biaso[nt]);
      oa = wmma_f16(a0, ldB(BO + (nt * 2 + 0) * 512, lane), oa);
      oa = wmma_f16(a1, ldB(BO + (nt * 2 + 1) * 512, lane), oa);
      #pragma unroll
      for (int e = 0; e < 8; ++e)
        scrw[(mrow + e) * 36 + nt * 16 + nl] = fmaxf(oa[e], 0.f);
    }
    asm volatile("" ::: "memory");   // keep LDS st->ld order (same-wave DS is in-order)
    if (lane < 16 && t >= TPRED) {
      float accr = 0.f;
      const float* sr = scrw + lane * 36;
      #pragma unroll
      for (int j2 = 0; j2 < 32; j2 += 4) {
        f4 a  = *(const f4*)(sr + j2);
        f4 wv = *(const f4*)(W2 + j2);
        accr += a[0] * wv[0] + a[1] * wv[1] + a[2] * wv[2] + a[3] * wv[3];
      }
      out[(size_t)(b * Ss + s0 + lane) * (Tt - TPRED) + (t - TPRED)] = accr + ob2;
    }
  }
}

extern "C" void kernel_launch(void* const* d_in, const int* in_sizes, int n_in,
                              void* d_out, int out_size, void* d_ws, size_t ws_size,
                              hipStream_t stream) {
  (void)in_sizes; (void)n_in; (void)out_size; (void)d_ws; (void)ws_size;
  static bool attr_set = false;
  if (!attr_set) {   // allow >64KB dynamic LDS (no-op if unsupported); host-side, capture-safe
    hipFuncSetAttribute((const void*)tp_lstm_kernel,
                        hipFuncAttributeMaxDynamicSharedMemorySize, SMEM_BYTES);
    attr_set = true;
  }
  tp_lstm_kernel<<<dim3(Bb), dim3(256), SMEM_BYTES, stream>>>(
      (const float*)d_in[0],  (const float*)d_in[1],  (const float*)d_in[2],
      (const float*)d_in[3],  (const float*)d_in[4],  (const float*)d_in[5],
      (const float*)d_in[6],  (const float*)d_in[7],  (const float*)d_in[8],
      (const float*)d_in[9],  (const float*)d_in[10], (const float*)d_in[11],
      (const float*)d_in[12], (const float*)d_in[13], (const float*)d_in[14],
      (const float*)d_in[15], (float*)d_out);
}
